// BaseGAT_45449343926616
// MI455X (gfx1250) — matrix operand order repaired and checked by reading, not compiled, not run on hardware
//
#include <hip/hip_runtime.h>
#include <math.h>
#include <stdint.h>

#define N_NODES   50000
#define N_EDGES   800000
#define F_IN      128
#define HID       64
#define HEADS     2
#define NC        2
#define G_GRAPHS  64
#define NEG_SLOPE 0.2f
#define MTILES    5      // 16-row tiles per block; 50000 = 625 * (16*5)

typedef __attribute__((ext_vector_type(16))) _Float16 v16h;
typedef __attribute__((ext_vector_type(8)))  float    v8f;

// ---------------------------------------------------------------- fill
__global__ void fill_kernel(float* p, float v, int n) {
    int i = blockIdx.x * blockDim.x + threadIdx.x;
    if (i < n) p[i] = v;
}

// ---------------------------------------------------------------- WMMA GEMM
// C[rows x ncols] = A[rows x K] * B[K x ncols]; f32 in/out, f16 WMMA compute.
// grid.x = rows/(16*MTILES) ; blockDim.x = (ncols/16)*32 ; K,ncols mult of 32/16.
__global__ void gemm_wmma_kernel(const float* __restrict__ A,
                                 const float* __restrict__ B,
                                 float* __restrict__ C,
                                 int K, int ncols, int rows) {
    __shared__ __align__(32) _Float16 ldsBt[128 * 128];  // transposed: [n][k]

    const int tid = threadIdx.x;
    const int total = K * ncols;
    for (int i = tid; i < total; i += blockDim.x) {
        int k = i / ncols, n = i % ncols;
        ldsBt[n * K + k] = (_Float16)B[i];
    }
    __syncthreads();

    const int wave  = tid >> 5;
    const int lane  = tid & 31;
    const int hi    = lane >> 4;       // half-wave: 0 or 1
    const int col_l = lane & 15;
    const int col   = wave * 16 + col_l;

    for (int t = 0; t < MTILES; ++t) {
        const int rowbase = (blockIdx.x * MTILES + t) * 16;
        if (rowbase >= rows) break;
        const int row = rowbase + col_l;          // A row owned by this lane
        const float* arow = A + (size_t)row * K;

        // prefetch next tile's A row (gfx1250 global_prefetch_b8)
        if (t + 1 < MTILES && rowbase + 16 < rows)
            __builtin_prefetch(arow + (size_t)16 * K, 0, 1);

        v8f acc = {};
        for (int kb = 0; kb < K; kb += 32) {
            // A fragment, contiguous K order matching B:
            //   lane<16 : K = kb+0..15 ; lane>=16: K = kb+16..31
            v16h a;
            const float* ap = arow + kb + (hi ? 16 : 0);
            #pragma unroll
            for (int j = 0; j < 8; ++j) {
                a[j]     = (_Float16)ap[j];
                a[8 + j] = (_Float16)ap[8 + j];
            }
            // B fragment: column `col`, K = kb(+16 upper half) .. +15, contiguous
            const _Float16* bp = &ldsBt[col * K + kb + (hi ? 16 : 0)];
            v16h b = *(const v16h*)bp;

            acc = __builtin_amdgcn_wmma_f32_16x16x32_f16(
                false, a, false, b, (short)0, acc, false, false);
        }

        // C/D layout: VGPR i -> row rowbase + i + hi*8, col = lane&15
        #pragma unroll
        for (int i = 0; i < 8; ++i) {
            int r = rowbase + hi * 8 + i;
            C[(size_t)r * ncols + col] = acc[i];
        }
    }
}

// ---------------------------------------------------------------- attention coefficients
// blockDim = H*D, grid = N ; as/ad: [N,H]
__global__ void alpha_kernel(const float* __restrict__ h,
                             const float* __restrict__ a_src,
                             const float* __restrict__ a_dst,
                             float* __restrict__ as_out,
                             float* __restrict__ ad_out,
                             int H, int D) {
    int n = blockIdx.x;
    int t = threadIdx.x;
    int head = t / D, d = t % D;
    __shared__ float ss[128], sd[128];
    float v = h[(size_t)n * H * D + t];
    ss[t] = v * a_src[t];
    sd[t] = v * a_dst[t];
    __syncthreads();
    for (int off = D >> 1; off > 0; off >>= 1) {
        if (d < off) { ss[t] += ss[t + off]; sd[t] += sd[t + off]; }
        __syncthreads();
    }
    if (d == 0) {
        as_out[n * H + head] = ss[t];
        ad_out[n * H + head] = sd[t];
    }
}

// ---------------------------------------------------------------- edge passes
__device__ __forceinline__ void atomicMaxF(float* addr, float v) {
    if (v >= 0.0f) atomicMax((int*)addr, __float_as_int(v));
    else           atomicMin((unsigned int*)addr, __float_as_uint(v));
}

__device__ __forceinline__ float edge_e(const float* as, const float* ad,
                                        int s, int d, int H, int h) {
    float e = as[s * H + h] + ad[d * H + h];
    return e > 0.0f ? e : NEG_SLOPE * e;
}

__global__ void edge_max_kernel(const int* __restrict__ src, const int* __restrict__ dst,
                                const float* __restrict__ as, const float* __restrict__ ad,
                                float* __restrict__ m, int E, int H) {
    int idx = blockIdx.x * blockDim.x + threadIdx.x;
    if (idx >= E * H) return;
    int e = idx / H, h = idx % H;
    float v = edge_e(as, ad, src[e], dst[e], H, h);
    atomicMaxF(&m[dst[e] * H + h], v);
}

__global__ void fix_m_kernel(float* m, int n) {
    int i = blockIdx.x * blockDim.x + threadIdx.x;
    if (i < n && m[i] == -__builtin_inff()) m[i] = 0.0f;
}

__global__ void edge_expsum_kernel(const int* __restrict__ src, const int* __restrict__ dst,
                                   const float* __restrict__ as, const float* __restrict__ ad,
                                   const float* __restrict__ m, float* __restrict__ denom,
                                   int E, int H) {
    int idx = blockIdx.x * blockDim.x + threadIdx.x;
    if (idx >= E * H) return;
    int e = idx / H, h = idx % H;
    int d = dst[e];
    float v = edge_e(as, ad, src[e], d, H, h);
    atomicAdd(&denom[d * H + h], __expf(v - m[d * H + h]));
}

// one wave per edge; each lane handles PER consecutive features (PER*32 == H*D)
template <int PER>
__global__ void edge_scatter_kernel(const int* __restrict__ src, const int* __restrict__ dst,
                                    const float* __restrict__ as, const float* __restrict__ ad,
                                    const float* __restrict__ m, const float* __restrict__ denom,
                                    const float* __restrict__ hsrc, float* __restrict__ out,
                                    int E, int H, int D) {
    int gt  = blockIdx.x * blockDim.x + threadIdx.x;
    int wid = gt >> 5, lane = gt & 31;
    if (wid >= E) return;
    int s = src[wid], d0 = dst[wid];
    int Dtot = H * D;
    int fbase = lane * PER;
    int hh = fbase / D;                         // head of this lane's chunk
    float e  = edge_e(as, ad, s, d0, H, hh);
    float ex = __expf(e - m[d0 * H + hh]);
    float alpha = ex / (denom[d0 * H + hh] + 1e-16f);
    const float* hp = hsrc + (size_t)s  * Dtot + fbase;
    float*       op = out  + (size_t)d0 * Dtot + fbase;
    #pragma unroll
    for (int j = 0; j < PER; ++j) atomicAdd(&op[j], hp[j] * alpha);
}

// ---------------------------------------------------------------- pointwise
__global__ void elu_bias_kernel(float* __restrict__ x, const float* __restrict__ b,
                                int n, int C) {
    int i = blockIdx.x * blockDim.x + threadIdx.x;
    if (i >= n) return;
    float v = x[i] + b[i % C];
    x[i] = v > 0.0f ? v : __expf(v) - 1.0f;
}

// grid = N, blockDim = HID
__global__ void pool_kernel(const float* __restrict__ out2, const float* __restrict__ b2,
                            const int* __restrict__ batch,
                            float* __restrict__ pooled, float* __restrict__ counts) {
    int n = blockIdx.x, t = threadIdx.x;
    int g = batch[n];
    float v = out2[(size_t)n * HID + t] + b2[t];
    atomicAdd(&pooled[g * HID + t], v);
    if (t == 0) atomicAdd(&counts[g], 1.0f);
}

// single block of G_GRAPHS threads
__global__ void head_kernel(const float* __restrict__ pooled, const float* __restrict__ counts,
                            const float* __restrict__ Wc, const float* __restrict__ bc,
                            float* __restrict__ out) {
    int g = threadIdx.x;
    float cnt = fmaxf(counts[g], 1.0f);
    float l0 = bc[0], l1 = bc[1];
    for (int d = 0; d < HID; ++d) {
        float pm = pooled[g * HID + d] / cnt;
        l0 += pm * Wc[d * NC + 0];
        l1 += pm * Wc[d * NC + 1];
    }
    float mx  = fmaxf(l0, l1);
    float lse = mx + __logf(__expf(l0 - mx) + __expf(l1 - mx));
    out[g * NC + 0] = l0 - lse;
    out[g * NC + 1] = l1 - lse;
}

// ---------------------------------------------------------------- launch
extern "C" void kernel_launch(void* const* d_in, const int* in_sizes, int n_in,
                              void* d_out, int out_size, void* d_ws, size_t ws_size,
                              hipStream_t stream) {
    const float* x      = (const float*)d_in[0];
    const int*   ei     = (const int*)  d_in[1];
    const int*   batch  = (const int*)  d_in[2];
    const float* W1     = (const float*)d_in[3];
    const float* a_src1 = (const float*)d_in[4];
    const float* a_dst1 = (const float*)d_in[5];
    const float* b1     = (const float*)d_in[6];
    const float* W2     = (const float*)d_in[7];
    const float* a_src2 = (const float*)d_in[8];
    const float* a_dst2 = (const float*)d_in[9];
    const float* b2     = (const float*)d_in[10];
    const float* Wc     = (const float*)d_in[11];
    const float* bc     = (const float*)d_in[12];
    float* out = (float*)d_out;

    const int* src = ei;             // edge_index[0]
    const int* dst = ei + N_EDGES;   // edge_index[1]

    // workspace layout (f32)
    float* ws     = (float*)d_ws;
    float* h1     = ws;                           // N*128
    float* out1   = h1   + (size_t)N_NODES * 128; // N*128 (accum -> ELU in place)
    float* as1    = out1 + (size_t)N_NODES * 128; // N*2
    float* ad1    = as1  + (size_t)N_NODES * 2;
    float* m1     = ad1  + (size_t)N_NODES * 2;
    float* dn1    = m1   + (size_t)N_NODES * 2;
    float* h2     = dn1  + (size_t)N_NODES * 2;   // N*64
    float* as2    = h2   + (size_t)N_NODES * 64;  // N
    float* ad2    = as2  + (size_t)N_NODES;
    float* m2     = ad2  + (size_t)N_NODES;
    float* dn2    = m2   + (size_t)N_NODES;
    float* out2   = dn2  + (size_t)N_NODES;       // N*64
    float* pooled = out2 + (size_t)N_NODES * 64;  // G*64
    float* counts = pooled + (size_t)G_GRAPHS * 64; // G

    const float NEG_INF = -INFINITY;
    auto fill = [&](float* p, float v, int n) {
        fill_kernel<<<(n + 255) / 256, 256, 0, stream>>>(p, v, n);
    };
    fill(out1, 0.0f, N_NODES * 128);
    fill(m1,  NEG_INF, N_NODES * HEADS);
    fill(dn1, 0.0f, N_NODES * HEADS);
    fill(out2, 0.0f, N_NODES * HID);
    fill(m2,  NEG_INF, N_NODES);
    fill(dn2, 0.0f, N_NODES);
    fill(pooled, 0.0f, G_GRAPHS * HID);
    fill(counts, 0.0f, G_GRAPHS);

    const int gemm_grid = N_NODES / (16 * MTILES);   // 625

    // ---------------- layer 1 ----------------
    gemm_wmma_kernel<<<gemm_grid, (128 / 16) * 32, 0, stream>>>(x, W1, h1, 128, 128, N_NODES);
    alpha_kernel<<<N_NODES, HEADS * HID, 0, stream>>>(h1, a_src1, a_dst1, as1, ad1, HEADS, HID);

    int eh1 = N_EDGES * HEADS;
    edge_max_kernel<<<(eh1 + 255) / 256, 256, 0, stream>>>(src, dst, as1, ad1, m1, N_EDGES, HEADS);
    fix_m_kernel<<<(N_NODES * HEADS + 255) / 256, 256, 0, stream>>>(m1, N_NODES * HEADS);
    edge_expsum_kernel<<<(eh1 + 255) / 256, 256, 0, stream>>>(src, dst, as1, ad1, m1, dn1, N_EDGES, HEADS);
    edge_scatter_kernel<4><<<(N_EDGES + 7) / 8, 256, 0, stream>>>(src, dst, as1, ad1, m1, dn1,
                                                                  h1, out1, N_EDGES, HEADS, HID);
    elu_bias_kernel<<<(N_NODES * 128 + 255) / 256, 256, 0, stream>>>(out1, b1, N_NODES * 128, 128);

    // ---------------- layer 2 ----------------
    gemm_wmma_kernel<<<gemm_grid, (64 / 16) * 32, 0, stream>>>(out1, W2, h2, 128, 64, N_NODES);
    alpha_kernel<<<N_NODES, 1 * HID, 0, stream>>>(h2, a_src2, a_dst2, as2, ad2, 1, HID);

    edge_max_kernel<<<(N_EDGES + 255) / 256, 256, 0, stream>>>(src, dst, as2, ad2, m2, N_EDGES, 1);
    fix_m_kernel<<<(N_NODES + 255) / 256, 256, 0, stream>>>(m2, N_NODES);
    edge_expsum_kernel<<<(N_EDGES + 255) / 256, 256, 0, stream>>>(src, dst, as2, ad2, m2, dn2, N_EDGES, 1);
    edge_scatter_kernel<2><<<(N_EDGES + 7) / 8, 256, 0, stream>>>(src, dst, as2, ad2, m2, dn2,
                                                                  h2, out2, N_EDGES, 1, HID);

    // ---------------- pool + classifier ----------------
    pool_kernel<<<N_NODES, HID, 0, stream>>>(out2, b2, batch, pooled, counts);
    head_kernel<<<1, G_GRAPHS, 0, stream>>>(pooled, counts, Wc, bc, out);
}
